// AdaptiveEnsembleGNN_86904368267867
// MI455X (gfx1250) — compile-verified, matrix operand order checked by reading, not run on hardware
//
#include <hip/hip_runtime.h>
#include <hip/hip_bf16.h>

// ---------------------------------------------------------------------------
// AdaptiveEnsembleGNN for MI455X (gfx1250, wave32, WMMA)
//   3x GIN conv (scatter-add + MLP w/ BatchNorm) + mean/max/add pooling +
//   3 classifiers + attention-softmax ensemble.
// GEMMs use v_wmma_f32_16x16x32_f16 (confirmed builtin), f32 accumulate.
// ---------------------------------------------------------------------------

#define HD   128          // hidden / input dim
#define CD   10           // classes
#define GD   1024         // graphs
#define BN_EPS 1e-5f

typedef __attribute__((ext_vector_type(16))) _Float16 v16h;
typedef __attribute__((ext_vector_type(8)))  float    v8f;
typedef __attribute__((ext_vector_type(4)))  float    v4f;

// ------------------------- small utility kernels ---------------------------

__global__ __launch_bounds__(256) void k_zero(float* p, int n) {
  int i = blockIdx.x * 256 + threadIdx.x;
  if (i < n) p[i] = 0.0f;
}

__global__ __launch_bounds__(256) void k_init_max(unsigned* p, int n) {
  int i = blockIdx.x * 256 + threadIdx.x;
  if (i < n) p[i] = 0x007FFFFFu;   // order-transform of -inf
}

__global__ __launch_bounds__(256) void k_embed(const int* __restrict__ x,
                                               const float* __restrict__ emb,
                                               float* __restrict__ h, int n) {
  int i = blockIdx.x * 256 + threadIdx.x;
  if (i >= n * HD) return;
  int node = i >> 7, d = i & (HD - 1);
  h[i] = emb[x[node] * HD + d];
}

// one wave per edge: agg[dst] += h[src]  (128 f32 per edge, 4 per lane)
__global__ __launch_bounds__(256) void k_scatter(const int* __restrict__ src,
                                                 const int* __restrict__ dst,
                                                 const float* __restrict__ h,
                                                 float* __restrict__ agg, int E) {
  int wid  = (blockIdx.x * 256 + threadIdx.x) >> 5;
  int lane = threadIdx.x & 31;
  if (wid >= E) return;
  int s = src[wid], d = dst[wid];
  v4f v = *(const v4f*)(h + (size_t)s * HD + lane * 4);
  float* p = agg + (size_t)d * HD + lane * 4;
  atomicAdd(p + 0, v.x);
  atomicAdd(p + 1, v.y);
  atomicAdd(p + 2, v.z);
  atomicAdd(p + 3, v.w);
}

// Pack f32 weight W[k][n] (128x128 row-major) into f16 WMMA-B-friendly layout:
//   Bp[((ks*2+hi)*128 + n)*16 + t]; t<8 -> K = ks*32 + hi*8 + t
//                                   t>=8 -> K = ks*32 + 16 + hi*8 + (t-8)
__global__ __launch_bounds__(256) void k_pack(const float* __restrict__ W,
                                              _Float16* __restrict__ Bp) {
  int i = blockIdx.x * 256 + threadIdx.x;       // 16384
  if (i >= HD * HD) return;
  int k = i >> 7, n = i & (HD - 1);
  int ks = k >> 5;
  int r  = k & 31;
  int half = (r >> 4) & 1;
  int hi   = (r >> 3) & 1;
  int t    = half * 8 + (r & 7);
  Bp[(((size_t)(ks * 2 + hi) * HD + n) << 4) + t] = (_Float16)W[(size_t)k * HD + n];
}

// ------------------------------ WMMA GEMMs ---------------------------------

__device__ __forceinline__ void cvt8_sum(v16h& a, int ofs,
                                         const float* p0, const float* p1) {
  v4f f0 = *(const v4f*)p0 + *(const v4f*)p1;
  v4f f1 = *(const v4f*)(p0 + 4) + *(const v4f*)(p1 + 4);
  a[ofs + 0] = (_Float16)f0.x; a[ofs + 1] = (_Float16)f0.y;
  a[ofs + 2] = (_Float16)f0.z; a[ofs + 3] = (_Float16)f0.w;
  a[ofs + 4] = (_Float16)f1.x; a[ofs + 5] = (_Float16)f1.y;
  a[ofs + 6] = (_Float16)f1.z; a[ofs + 7] = (_Float16)f1.w;
}

// pre = (h + agg) @ W + bias    one wave: 16 rows x 128 cols
__global__ __launch_bounds__(256) void k_gemm_sum(
    const float* __restrict__ h, const float* __restrict__ agg,
    const _Float16* __restrict__ Bp, const float* __restrict__ bias,
    float* __restrict__ out, int M) {
  int wid  = (blockIdx.x * 256 + threadIdx.x) >> 5;
  int lane = threadIdx.x & 31;
  int row0 = wid * 16;
  if (row0 >= M) return;
  int lr = lane & 15, hi = lane >> 4;
  const float* hrow = h   + (size_t)(row0 + lr) * HD;
  const float* arow = agg + (size_t)(row0 + lr) * HD;
  const v16h*  bpv  = (const v16h*)Bp;

  v8f acc[8];
  v8f z = {};
#pragma unroll
  for (int t = 0; t < 8; ++t) acc[t] = z;

#pragma unroll
  for (int ks = 0; ks < 4; ++ks) {
    int kb0 = ks * 32 + hi * 8;
    v16h a;
    cvt8_sum(a, 0, hrow + kb0,      arow + kb0);
    cvt8_sum(a, 8, hrow + kb0 + 16, arow + kb0 + 16);
    int bb = (ks * 2 + hi) * HD + lr;
#pragma unroll
    for (int nt = 0; nt < 8; ++nt) {
      v16h b = bpv[bb + nt * 16];
      acc[nt] = __builtin_amdgcn_wmma_f32_16x16x32_f16(
          false, a, false, b, (short)0, acc[nt], false, false);
    }
  }
#pragma unroll
  for (int nt = 0; nt < 8; ++nt) {
    int col = nt * 16 + lr;
    float bv = bias[col];
    float* orow = out + (size_t)(row0 + hi * 8) * HD + col;
#pragma unroll
    for (int r = 0; r < 8; ++r) orow[(size_t)r * HD] = acc[nt][r] + bv;
  }
}

// hout = ReLU((pre-mu)*rs + bt) @ W + bias (+ res)   (BN fused into A-load)
__global__ __launch_bounds__(256) void k_gemm_bn(
    const float* __restrict__ pre, const float* __restrict__ mu,
    const float* __restrict__ rs,  const float* __restrict__ bt,
    const _Float16* __restrict__ Bp, const float* __restrict__ bias,
    const float* __restrict__ res, float* __restrict__ out, int M) {
  int wid  = (blockIdx.x * 256 + threadIdx.x) >> 5;
  int lane = threadIdx.x & 31;
  int row0 = wid * 16;
  if (row0 >= M) return;
  int lr = lane & 15, hi = lane >> 4;
  const float* prow = pre + (size_t)(row0 + lr) * HD;
  const v16h*  bpv  = (const v16h*)Bp;

  v8f acc[8];
  v8f z = {};
#pragma unroll
  for (int t = 0; t < 8; ++t) acc[t] = z;

#pragma unroll
  for (int ks = 0; ks < 4; ++ks) {
    int kb0 = ks * 32 + hi * 8;
    v16h a;
#pragma unroll
    for (int q = 0; q < 8; ++q) {
      int k0 = kb0 + q, k1 = kb0 + 16 + q;
      float f0 = (prow[k0] - mu[k0]) * rs[k0] + bt[k0];
      float f1 = (prow[k1] - mu[k1]) * rs[k1] + bt[k1];
      a[q]     = (_Float16)fmaxf(f0, 0.0f);
      a[q + 8] = (_Float16)fmaxf(f1, 0.0f);
    }
    int bb = (ks * 2 + hi) * HD + lr;
#pragma unroll
    for (int nt = 0; nt < 8; ++nt) {
      v16h b = bpv[bb + nt * 16];
      acc[nt] = __builtin_amdgcn_wmma_f32_16x16x32_f16(
          false, a, false, b, (short)0, acc[nt], false, false);
    }
  }
#pragma unroll
  for (int nt = 0; nt < 8; ++nt) {
    int col = nt * 16 + lr;
    float bv = bias[col];
    float* orow = out + (size_t)(row0 + hi * 8) * HD + col;
    const float* rrow = res ? res + (size_t)(row0 + hi * 8) * HD + col : nullptr;
#pragma unroll
    for (int r = 0; r < 8; ++r) {
      float v = acc[nt][r] + bv;
      if (res) v += rrow[(size_t)r * HD];
      orow[(size_t)r * HD] = v;
    }
  }
}

// ---------------------------- BatchNorm stats ------------------------------

__global__ __launch_bounds__(128) void k_colstats(const float* __restrict__ pre,
                                                  float* __restrict__ csum,
                                                  float* __restrict__ csq, int M) {
  int c  = threadIdx.x;                   // 128 columns
  int r0 = blockIdx.x * 256;
  int r1 = min(r0 + 256, M);
  float s = 0.0f, q = 0.0f;
  for (int r = r0; r < r1; ++r) {
    float v = pre[(size_t)r * HD + c];
    s += v; q += v * v;
  }
  atomicAdd(&csum[c], s);
  atomicAdd(&csq[c], q);
}

__global__ __launch_bounds__(128) void k_bnfinal(const float* __restrict__ csum,
                                                 const float* __restrict__ csq,
                                                 const float* __restrict__ g,
                                                 float* __restrict__ mu,
                                                 float* __restrict__ rs, int M) {
  int c = threadIdx.x;
  if (c >= HD) return;
  float m = csum[c] / (float)M;
  float v = csq[c] / (float)M - m * m;
  mu[c] = m;
  rs[c] = g[c] * rsqrtf(v + BN_EPS);
}

// ------------------------------- pooling -----------------------------------

__device__ __forceinline__ unsigned f2o(float f) {
  unsigned b = __float_as_uint(f);
  return (b & 0x80000000u) ? ~b : (b | 0x80000000u);
}
__device__ __forceinline__ float o2f(unsigned u) {
  return (u & 0x80000000u) ? __uint_as_float(u & 0x7FFFFFFFu)
                           : __uint_as_float(~u);
}

// one wave per node: add_r/max_r/counts
__global__ __launch_bounds__(256) void k_pool(const float* __restrict__ h3,
                                              const int* __restrict__ batch,
                                              float* __restrict__ add_r,
                                              unsigned* __restrict__ max_u,
                                              float* __restrict__ counts, int n) {
  int wid  = (blockIdx.x * 256 + threadIdx.x) >> 5;
  int lane = threadIdx.x & 31;
  if (wid >= n) return;
  int g = batch[wid];
  v4f v = *(const v4f*)(h3 + (size_t)wid * HD + lane * 4);
  float* pa = add_r + (size_t)g * HD + lane * 4;
  atomicAdd(pa + 0, v.x); atomicAdd(pa + 1, v.y);
  atomicAdd(pa + 2, v.z); atomicAdd(pa + 3, v.w);
  unsigned* pm = max_u + (size_t)g * HD + lane * 4;
  atomicMax(pm + 0, f2o(v.x)); atomicMax(pm + 1, f2o(v.y));
  atomicMax(pm + 2, f2o(v.z)); atomicMax(pm + 3, f2o(v.w));
  if (lane == 0) atomicAdd(&counts[g], 1.0f);
}

// ------------------------------ graph head ---------------------------------
// one block (128 threads) per graph: readouts -> 3 classifiers + attention MLP
__global__ __launch_bounds__(128) void k_head(
    const float* __restrict__ add_r, const unsigned* __restrict__ max_u,
    const float* __restrict__ counts,
    const float* __restrict__ Wm, const float* __restrict__ bm,
    const float* __restrict__ Wx, const float* __restrict__ bx,
    const float* __restrict__ Wa, const float* __restrict__ ba,
    const float* __restrict__ W1, const float* __restrict__ b1,
    const float* __restrict__ W2, const float* __restrict__ b2,
    float* __restrict__ o_ens, float* __restrict__ o_lm,
    float* __restrict__ o_lx,  float* __restrict__ o_la,
    float* __restrict__ o_attn) {
  int g = blockIdx.x, c = threadIdx.x;
  __shared__ float comb[3 * HD];
  __shared__ float ah[HD];
  __shared__ float at[3];

  float cnt  = counts[g];
  float addv = add_r[(size_t)g * HD + c];
  float meanv = addv / fmaxf(cnt, 1.0f);
  float maxv  = (cnt > 0.0f) ? o2f(max_u[(size_t)g * HD + c]) : 0.0f;
  comb[c] = meanv; comb[HD + c] = maxv; comb[2 * HD + c] = addv;
  __syncthreads();

  // attention hidden: [384] @ [384,128]
  float s = b1[c];
  for (int k = 0; k < 3 * HD; ++k) s += comb[k] * W1[(size_t)k * HD + c];
  ah[c] = fmaxf(s, 0.0f);
  __syncthreads();

  float sm = 0.0f, sx = 0.0f, sa = 0.0f;
  if (c < CD) {
    sm = bm[c]; sx = bx[c]; sa = ba[c];
    for (int d = 0; d < HD; ++d) {
      sm += comb[d] * Wm[(size_t)d * CD + c];
      sx += comb[HD + d] * Wx[(size_t)d * CD + c];
      sa += comb[2 * HD + d] * Wa[(size_t)d * CD + c];
    }
  }
  if (c >= 16 && c < 19) {
    int j = c - 16;
    float s2 = b2[j];
    for (int d = 0; d < HD; ++d) s2 += ah[d] * W2[(size_t)d * 3 + j];
    at[j] = s2;
  }
  __syncthreads();

  if (c == 0) {
    float m = fmaxf(at[0], fmaxf(at[1], at[2]));
    float e0 = __expf(at[0] - m), e1 = __expf(at[1] - m), e2 = __expf(at[2] - m);
    float inv = 1.0f / (e0 + e1 + e2);
    at[0] = e0 * inv; at[1] = e1 * inv; at[2] = e2 * inv;
  }
  __syncthreads();

  if (c < CD) {
    o_lm[(size_t)g * CD + c] = sm;
    o_lx[(size_t)g * CD + c] = sx;
    o_la[(size_t)g * CD + c] = sa;
    o_ens[(size_t)g * CD + c] = at[0] * sm + at[1] * sx + at[2] * sa;
  }
  if (c < 3) o_attn[(size_t)g * 3 + c] = at[c];
}

// ------------------------------- launcher ----------------------------------

static inline int cdiv(long a, long b) { return (int)((a + b - 1) / b); }

extern "C" void kernel_launch(void* const* d_in, const int* in_sizes, int n_in,
                              void* d_out, int out_size, void* d_ws, size_t ws_size,
                              hipStream_t stream) {
  const int N = in_sizes[0];
  const int E = in_sizes[1] / 2;

  const int*   x     = (const int*)d_in[0];
  const int*   ei    = (const int*)d_in[1];
  const int*   src   = ei;
  const int*   dst   = ei + E;
  const int*   batch = (const int*)d_in[2];
  const float* emb   = (const float*)d_in[3];

  struct ConvP { const float *w1, *b1, *g, *bt, *w2, *b2; };
  ConvP cp[3];
  for (int l = 0; l < 3; ++l) {
    int b = 4 + l * 6;
    cp[l].w1 = (const float*)d_in[b + 0];
    cp[l].b1 = (const float*)d_in[b + 1];
    cp[l].g  = (const float*)d_in[b + 2];
    cp[l].bt = (const float*)d_in[b + 3];
    cp[l].w2 = (const float*)d_in[b + 4];
    cp[l].b2 = (const float*)d_in[b + 5];
  }
  const float* Wm = (const float*)d_in[22];
  const float* bm = (const float*)d_in[23];
  const float* Wx = (const float*)d_in[24];
  const float* bx = (const float*)d_in[25];
  const float* Wa = (const float*)d_in[26];
  const float* ba = (const float*)d_in[27];
  const float* W1 = (const float*)d_in[28];
  const float* b1 = (const float*)d_in[29];
  const float* W2 = (const float*)d_in[30];
  const float* b2 = (const float*)d_in[31];

  // workspace layout
  char* ws = (char*)d_ws;
  size_t hbytes = (size_t)N * HD * sizeof(float);
  float* hA  = (float*)(ws);
  float* hB  = (float*)(ws + hbytes);
  float* agg = (float*)(ws + 2 * hbytes);
  float* pre = (float*)(ws + 3 * hbytes);
  char*  sm  = ws + 4 * hbytes;
  _Float16* Bp = (_Float16*)sm;                 sm += 32768;
  float* colsum   = (float*)sm;                 sm += 128 * sizeof(float);
  float* colsumsq = (float*)sm;                 sm += 128 * sizeof(float);
  float* mu       = (float*)sm;                 sm += 128 * sizeof(float);
  float* rs       = (float*)sm;                 sm += 128 * sizeof(float);
  float* add_r    = (float*)sm;                 sm += (size_t)GD * HD * sizeof(float);
  float* counts   = (float*)sm;                 sm += (size_t)GD * sizeof(float);
  unsigned* max_u = (unsigned*)sm;              sm += (size_t)GD * HD * sizeof(unsigned);

  float* out   = (float*)d_out;
  float* o_ens = out;
  float* o_lm  = out + (size_t)GD * CD;
  float* o_lx  = out + 2 * (size_t)GD * CD;
  float* o_la  = out + 3 * (size_t)GD * CD;
  float* o_at  = out + 4 * (size_t)GD * CD;

  const int nElem  = N * HD;
  const int gWaves = cdiv(N, 16);
  const int gBlks  = cdiv((long)gWaves * 32, 256);

  // h0 = emb[x]
  k_embed<<<cdiv(nElem, 256), 256, 0, stream>>>(x, emb, hA, N);

  float* hin = hA;
  float* hout = hB;
  for (int l = 0; l < 3; ++l) {
    k_zero<<<cdiv(nElem, 256), 256, 0, stream>>>(agg, nElem);
    k_scatter<<<cdiv((long)E * 32, 256), 256, 0, stream>>>(src, dst, hin, agg, E);
    k_pack<<<cdiv(HD * HD, 256), 256, 0, stream>>>(cp[l].w1, Bp);
    k_gemm_sum<<<gBlks, 256, 0, stream>>>(hin, agg, Bp, cp[l].b1, pre, N);
    k_zero<<<1, 256, 0, stream>>>(colsum, 256);            // colsum + colsumsq
    k_colstats<<<cdiv(N, 256), 128, 0, stream>>>(pre, colsum, colsumsq, N);
    k_bnfinal<<<1, 128, 0, stream>>>(colsum, colsumsq, cp[l].g, mu, rs, N);
    k_pack<<<cdiv(HD * HD, 256), 256, 0, stream>>>(cp[l].w2, Bp);
    const float* res = (l > 0) ? hin : nullptr;
    k_gemm_bn<<<gBlks, 256, 0, stream>>>(pre, mu, rs, cp[l].bt, Bp, cp[l].b2,
                                         res, hout, N);
    float* t = hin; hin = hout; hout = t;
  }
  // hin now holds h3

  k_zero<<<cdiv((long)GD * (HD + 1), 256), 256, 0, stream>>>(add_r, GD * (HD + 1));
  k_init_max<<<cdiv((long)GD * HD, 256), 256, 0, stream>>>(max_u, GD * HD);
  k_pool<<<cdiv((long)N * 32, 256), 256, 0, stream>>>(hin, batch, add_r, max_u,
                                                      counts, N);
  k_head<<<GD, 128, 0, stream>>>(add_r, max_u, counts, Wm, bm, Wx, bx, Wa, ba,
                                 W1, b1, W2, b2, o_ens, o_lm, o_lx, o_la, o_at);

  (void)n_in; (void)out_size; (void)ws_size;
}
